// Encoder_38689065402592
// MI455X (gfx1250) — matrix operand order repaired and checked by reading, not compiled, not run on hardware
//
#include <hip/hip_runtime.h>
#include <hip/hip_bf16.h>

typedef __attribute__((ext_vector_type(16))) _Float16 v16h;
typedef __attribute__((ext_vector_type(8)))  _Float16 v8h;
typedef __attribute__((ext_vector_type(8)))  float    v8f;

#define HID  64
#define CIN  1
#define COUT 256
#define BB   16
#define TT   12
#define HH   64
#define WW   64

static __device__ __forceinline__ v16h cat8(v8h lo, v8h hi) {
    v16h r;
#pragma unroll
    for (int i = 0; i < 8; ++i) { r[i] = lo[i]; r[i + 8] = hi[i]; }
    return r;
}

static __device__ __forceinline__ float sigmoidf_(float x) {
    return 1.0f / (1.0f + __expf(-x));
}

// ---------------------------------------------------------------------------
// Prep: repack weights to f16, WMMA-A-friendly (K = hidden channel contiguous)
//   Wpack [g][c][tap][hc]  (3*64*9*64 f16)   hidden-channel part of Wz/Wr/Wh
//   Wxpack[g][tap][c]      (3*9*64 f32)      x-channel part
//   Wopack[o][c]           (256*64 f16)      1x1 projection
// ---------------------------------------------------------------------------
__global__ void prep_pack(const float* __restrict__ Wz, const float* __restrict__ Wr,
                          const float* __restrict__ Wh, const float* __restrict__ Wo,
                          _Float16* __restrict__ Wpack, float* __restrict__ Wxpack,
                          _Float16* __restrict__ Wopack)
{
    int idx = blockIdx.x * blockDim.x + threadIdx.x;
    if (idx < 3 * 64 * 9 * 64) {
        int hc  = idx & 63;
        int tap = (idx >> 6) % 9;
        int c   = (idx / (64 * 9)) & 63;
        int g   = idx / (64 * 9 * 64);
        const float* Wg = (g == 0) ? Wz : (g == 1) ? Wr : Wh;
        float w = Wg[((c * (HID + CIN) + (CIN + hc)) * 3 + tap / 3) * 3 + (tap % 3)];
        Wpack[idx] = (_Float16)w;
    }
    if (idx < 3 * 9 * 64) {
        int c   = idx & 63;
        int tap = (idx >> 6) % 9;
        int g   = idx / (64 * 9);
        const float* Wg = (g == 0) ? Wz : (g == 1) ? Wr : Wh;
        Wxpack[idx] = Wg[((c * (HID + CIN) + 0) * 3 + tap / 3) * 3 + (tap % 3)];
    }
    if (idx < COUT * HID) {
        Wopack[idx] = (_Float16)Wo[idx];
    }
}

// ---------------------------------------------------------------------------
// Kernel 1: gates.  One workgroup = one (b, y) image row.
// 4 waves, wave w computes output channels [16w,16w+16) for all 64 pixels.
// z = sigmoid(conv3x3([x,h],Wz)+bz);  r likewise;  stores z and r*h (f16, [x][c])
// ---------------------------------------------------------------------------
__global__ void __launch_bounds__(128)
gru_gates(const float* __restrict__ x,
          const _Float16* __restrict__ Wpack,
          const float* __restrict__ Wxpack,
          const float* __restrict__ bz, const float* __restrict__ br,
          const _Float16* __restrict__ enc,      // h_{t-1} lives at enc[t-1]
          _Float16* __restrict__ zbuf, _Float16* __restrict__ rhbuf,
          int t)
{
    __shared__ __align__(16) _Float16 sh_h[3 * 66 * 64];   // [row][xm][c]
    __shared__ float sh_x[3 * 66];
    __shared__ __align__(16) _Float16 sh_z[64 * 64];       // [x][c]
    __shared__ __align__(16) _Float16 sh_rh[64 * 64];

    const int b = blockIdx.y, y = blockIdx.x;
    const int tid = threadIdx.x;
    const int lane = tid & 31, wave = tid >> 5;

    // ---- stage h_{t-1} rows y-1..y+1 (channel-contiguous, zero pad) ----
    const _Float16* hprev = enc + (size_t)(b * TT + (t - 1)) * HH * WW * HID;
    for (int i = tid; i < 3 * 64 * 32; i += 128) {
        int c2 = i & 31;
        int xx = (i >> 5) & 63;
        int r  = i >> 11;
        int yy = y - 1 + r;
        unsigned v = 0u;
        if (t > 0 && yy >= 0 && yy < HH)
            v = ((const unsigned*)hprev)[((size_t)yy * WW + xx) * 32 + c2];
        ((unsigned*)sh_h)[((r * 66) + (xx + 1)) * 32 + c2] = v;
    }
    for (int i = tid; i < 3 * 2 * 32; i += 128) {     // x-pad columns
        int c2 = i & 31;
        int p  = (i >> 5) & 1;
        int r  = i >> 6;
        ((unsigned*)sh_h)[((r * 66) + (p ? 65 : 0)) * 32 + c2] = 0u;
    }
    // ---- stage x rows ----
    const float* xp = x + (size_t)(b * TT + t) * HH * WW;
    for (int i = tid; i < 3 * 66; i += 128) {
        int xm = i % 66, r = i / 66;
        int yy = y - 1 + r, xx = xm - 1;
        float v = 0.f;
        if (yy >= 0 && yy < HH && xx >= 0 && xx < WW) v = xp[yy * WW + xx];
        sh_x[r * 66 + xm] = v;
    }
    __syncthreads();

    const int c0   = wave * 16;
    const int rowA = lane & 15;
    const int hiA  = (lane >> 4) & 1;

    v8f accz[4], accr[4];
#pragma unroll
    for (int p = 0; p < 4; ++p) { accz[p] = {}; accr[p] = {}; }

#pragma unroll
    for (int ty = 0; ty < 3; ++ty)
#pragma unroll
    for (int tx = 0; tx < 3; ++tx) {
        const int tap = ty * 3 + tx;
        v16h az[2], ar[2];
#pragma unroll
        for (int kk = 0; kk < 2; ++kk) {
            int hcb = kk * 32 + hiA * 8;
            const v8h* pz = (const v8h*)(Wpack + (((0 * 64 + (c0 + rowA)) * 9 + tap) * 64 + hcb));
            const v8h* pr = (const v8h*)(Wpack + (((1 * 64 + (c0 + rowA)) * 9 + tap) * 64 + hcb));
            az[kk] = cat8(pz[0], pz[2]);
            ar[kk] = cat8(pr[0], pr[2]);
        }
#pragma unroll
        for (int kk = 0; kk < 2; ++kk) {
#pragma unroll
            for (int pt = 0; pt < 4; ++pt) {
                int px = pt * 16 + rowA;
                const v8h* pb = (const v8h*)(sh_h + ((ty * 66 + (px + tx)) * 64 + kk * 32 + hiA * 8));
                v16h bf = cat8(pb[0], pb[2]);
                accz[pt] = __builtin_amdgcn_wmma_f32_16x16x32_f16(false, az[kk], false, bf,
                                                                  (short)0, accz[pt], false, false);
                accr[pt] = __builtin_amdgcn_wmma_f32_16x16x32_f16(false, ar[kk], false, bf,
                                                                  (short)0, accr[pt], false, false);
            }
        }
        // x-channel (CIN=1) contribution via VALU FMAs
#pragma unroll
        for (int v = 0; v < 8; ++v) {
            int c = c0 + v + hiA * 8;
            float wxz = Wxpack[(0 * 9 + tap) * 64 + c];
            float wxr = Wxpack[(1 * 9 + tap) * 64 + c];
#pragma unroll
            for (int pt = 0; pt < 4; ++pt) {
                float xv = sh_x[ty * 66 + pt * 16 + rowA + tx];
                accz[pt][v] += wxz * xv;
                accr[pt][v] += wxr * xv;
            }
        }
    }

    // epilogue: sigmoid, r*h, stage into LDS for coalesced writeout
#pragma unroll
    for (int pt = 0; pt < 4; ++pt) {
#pragma unroll
        for (int v = 0; v < 8; ++v) {
            int c  = c0 + v + hiA * 8;
            int px = pt * 16 + rowA;
            float zv = sigmoidf_(accz[pt][v] + bz[c]);
            float rv = sigmoidf_(accr[pt][v] + br[c]);
            float hv = (float)sh_h[(1 * 66 + (px + 1)) * 64 + c];
            sh_z[px * 64 + c]  = (_Float16)zv;
            sh_rh[px * 64 + c] = (_Float16)(rv * hv);
        }
    }
    __syncthreads();

    _Float16* zdst = zbuf  + (size_t)(b * HH + y) * WW * HID;
    _Float16* rdst = rhbuf + (size_t)(b * HH + y) * WW * HID;
    for (int i = tid; i < 64 * 32; i += 128) {
        ((unsigned*)zdst)[i] = ((const unsigned*)sh_z)[i];
        ((unsigned*)rdst)[i] = ((const unsigned*)sh_rh)[i];
    }
}

// ---------------------------------------------------------------------------
// Kernel 2: candidate + GRU update. Same tiling as kernel 1.
// cand = tanh(conv3x3([x, r*h], Wh)+bh);  h_t = (1-z)*h + z*cand -> enc[t]
// ---------------------------------------------------------------------------
__global__ void __launch_bounds__(128)
gru_cand(const float* __restrict__ x,
         const _Float16* __restrict__ Wpack,
         const float* __restrict__ Wxpack,
         const float* __restrict__ bh,
         _Float16* __restrict__ enc,
         const _Float16* __restrict__ zbuf, const _Float16* __restrict__ rhbuf,
         int t)
{
    __shared__ __align__(16) _Float16 sh_rh3[3 * 66 * 64];
    __shared__ float sh_x[3 * 66];
    __shared__ __align__(16) _Float16 sh_z[64 * 64];
    __shared__ __align__(16) _Float16 sh_h[64 * 64];
    __shared__ __align__(16) _Float16 sh_o[64 * 64];

    const int b = blockIdx.y, y = blockIdx.x;
    const int tid = threadIdx.x;
    const int lane = tid & 31, wave = tid >> 5;

    // stage r*h rows y-1..y+1
    for (int i = tid; i < 3 * 64 * 32; i += 128) {
        int c2 = i & 31;
        int xx = (i >> 5) & 63;
        int r  = i >> 11;
        int yy = y - 1 + r;
        unsigned v = 0u;
        if (yy >= 0 && yy < HH)
            v = ((const unsigned*)(rhbuf + (size_t)(b * HH + yy) * WW * HID))[(size_t)xx * 32 + c2];
        ((unsigned*)sh_rh3)[((r * 66) + (xx + 1)) * 32 + c2] = v;
    }
    for (int i = tid; i < 3 * 2 * 32; i += 128) {
        int c2 = i & 31;
        int p  = (i >> 5) & 1;
        int r  = i >> 6;
        ((unsigned*)sh_rh3)[((r * 66) + (p ? 65 : 0)) * 32 + c2] = 0u;
    }
    // stage x rows
    const float* xp = x + (size_t)(b * TT + t) * HH * WW;
    for (int i = tid; i < 3 * 66; i += 128) {
        int xm = i % 66, r = i / 66;
        int yy = y - 1 + r, xx = xm - 1;
        float v = 0.f;
        if (yy >= 0 && yy < HH && xx >= 0 && xx < WW) v = xp[yy * WW + xx];
        sh_x[r * 66 + xm] = v;
    }
    // stage z (center row) and h_{t-1} (center row)
    {
        const unsigned* zsrc = (const unsigned*)(zbuf + (size_t)(b * HH + y) * WW * HID);
        const _Float16* hp = enc + ((size_t)(b * TT + (t - 1)) * HH + y) * WW * HID;
        for (int i = tid; i < 64 * 32; i += 128) {
            ((unsigned*)sh_z)[i] = zsrc[i];
            ((unsigned*)sh_h)[i] = (t > 0) ? ((const unsigned*)hp)[i] : 0u;
        }
    }
    __syncthreads();

    const int c0   = wave * 16;
    const int rowA = lane & 15;
    const int hiA  = (lane >> 4) & 1;

    v8f acc[4];
#pragma unroll
    for (int p = 0; p < 4; ++p) acc[p] = {};

#pragma unroll
    for (int ty = 0; ty < 3; ++ty)
#pragma unroll
    for (int tx = 0; tx < 3; ++tx) {
        const int tap = ty * 3 + tx;
        v16h ah[2];
#pragma unroll
        for (int kk = 0; kk < 2; ++kk) {
            int hcb = kk * 32 + hiA * 8;
            const v8h* ph = (const v8h*)(Wpack + (((2 * 64 + (c0 + rowA)) * 9 + tap) * 64 + hcb));
            ah[kk] = cat8(ph[0], ph[2]);
        }
#pragma unroll
        for (int kk = 0; kk < 2; ++kk) {
#pragma unroll
            for (int pt = 0; pt < 4; ++pt) {
                int px = pt * 16 + rowA;
                const v8h* pb = (const v8h*)(sh_rh3 + ((ty * 66 + (px + tx)) * 64 + kk * 32 + hiA * 8));
                v16h bf = cat8(pb[0], pb[2]);
                acc[pt] = __builtin_amdgcn_wmma_f32_16x16x32_f16(false, ah[kk], false, bf,
                                                                 (short)0, acc[pt], false, false);
            }
        }
#pragma unroll
        for (int v = 0; v < 8; ++v) {
            int c = c0 + v + hiA * 8;
            float wxh = Wxpack[(2 * 9 + tap) * 64 + c];
#pragma unroll
            for (int pt = 0; pt < 4; ++pt) {
                float xv = sh_x[ty * 66 + pt * 16 + rowA + tx];
                acc[pt][v] += wxh * xv;
            }
        }
    }

#pragma unroll
    for (int pt = 0; pt < 4; ++pt) {
#pragma unroll
        for (int v = 0; v < 8; ++v) {
            int c  = c0 + v + hiA * 8;
            int px = pt * 16 + rowA;
            float cand = tanhf(acc[pt][v] + bh[c]);
            float zv = (float)sh_z[px * 64 + c];
            float hv = (float)sh_h[px * 64 + c];
            float hn = (1.0f - zv) * hv + zv * cand;
            sh_o[px * 64 + c] = (_Float16)hn;
        }
    }
    __syncthreads();

    _Float16* hdst = enc + ((size_t)(b * TT + t) * HH + y) * WW * HID;
    for (int i = tid; i < 64 * 32; i += 128)
        ((unsigned*)hdst)[i] = ((const unsigned*)sh_o)[i];
}

// ---------------------------------------------------------------------------
// Kernel 3: 1x1 projection GEMM.  out[bt,o,y,x] = sum_c Wo[o,c]*enc[bt,y,x,c]+bo
// grid: (4 out-ch blocks of 64) x (64 rows) x (192 frames); 4 waves/WG.
// ---------------------------------------------------------------------------
__global__ void __launch_bounds__(128)
proj(const _Float16* __restrict__ enc, const _Float16* __restrict__ Wopack,
     const float* __restrict__ bo, float* __restrict__ out)
{
    __shared__ __align__(16) _Float16 sh_e[64 * 64];

    const int ocb = blockIdx.x, y = blockIdx.y, bt = blockIdx.z;
    const int tid = threadIdx.x;
    const int lane = tid & 31, wave = tid >> 5;

    const unsigned* esrc = (const unsigned*)(enc + ((size_t)bt * HH + y) * WW * HID);
    for (int i = tid; i < 64 * 32; i += 128)
        ((unsigned*)sh_e)[i] = esrc[i];
    __syncthreads();

    const int c0   = ocb * 64 + wave * 16;
    const int rowA = lane & 15;
    const int hiA  = (lane >> 4) & 1;

    v16h a[2];
#pragma unroll
    for (int kk = 0; kk < 2; ++kk) {
        int kb = kk * 32 + hiA * 8;
        const v8h* pa = (const v8h*)(Wopack + ((c0 + rowA) * 64 + kb));
        a[kk] = cat8(pa[0], pa[2]);
    }

    v8f acc[4];
#pragma unroll
    for (int p = 0; p < 4; ++p) acc[p] = {};

#pragma unroll
    for (int kk = 0; kk < 2; ++kk) {
#pragma unroll
        for (int pt = 0; pt < 4; ++pt) {
            const v8h* pb = (const v8h*)(sh_e + ((pt * 16 + rowA) * 64 + kk * 32 + hiA * 8));
            v16h bf = cat8(pb[0], pb[2]);
            acc[pt] = __builtin_amdgcn_wmma_f32_16x16x32_f16(false, a[kk], false, bf,
                                                             (short)0, acc[pt], false, false);
        }
    }

#pragma unroll
    for (int pt = 0; pt < 4; ++pt) {
#pragma unroll
        for (int v = 0; v < 8; ++v) {
            int o  = c0 + v + hiA * 8;
            int px = pt * 16 + rowA;
            out[((size_t)bt * COUT + o) * (HH * WW) + y * WW + px] = acc[pt][v] + bo[o];
        }
    }
}

// ---------------------------------------------------------------------------
extern "C" void kernel_launch(void* const* d_in, const int* in_sizes, int n_in,
                              void* d_out, int out_size, void* d_ws, size_t ws_size,
                              hipStream_t stream)
{
    const float* x  = (const float*)d_in[0];
    const float* Wz = (const float*)d_in[1];
    const float* bz = (const float*)d_in[2];
    const float* Wr = (const float*)d_in[3];
    const float* br = (const float*)d_in[4];
    const float* Wh = (const float*)d_in[5];
    const float* bh = (const float*)d_in[6];
    const float* Wo = (const float*)d_in[7];
    const float* bo = (const float*)d_in[8];
    float* out = (float*)d_out;

    // workspace layout (all offsets 256B-aligned by construction)
    const size_t offWpack = 0;
    const size_t szWpack  = (size_t)3 * 64 * 9 * 64 * sizeof(_Float16);   // 221184
    const size_t offWx    = offWpack + szWpack;
    const size_t szWx     = (size_t)3 * 9 * 64 * sizeof(float);           // 6912
    const size_t offWo    = (offWx + szWx + 255) & ~(size_t)255;
    const size_t szWo     = (size_t)COUT * HID * sizeof(_Float16);        // 32768
    const size_t offEnc   = (offWo + szWo + 255) & ~(size_t)255;
    const size_t szEnc    = (size_t)BB * TT * HH * WW * HID * sizeof(_Float16);
    const size_t offZ     = offEnc + szEnc;
    const size_t szRow    = (size_t)BB * HH * WW * HID * sizeof(_Float16);
    const size_t offRH    = offZ + szRow;
    const size_t need     = offRH + szRow;
    if (d_ws == nullptr || ws_size < need) return;

    char* ws = (char*)d_ws;
    _Float16* Wpack  = (_Float16*)(ws + offWpack);
    float*    Wxpack = (float*)(ws + offWx);
    _Float16* Wopack = (_Float16*)(ws + offWo);
    _Float16* enc    = (_Float16*)(ws + offEnc);
    _Float16* zbuf   = (_Float16*)(ws + offZ);
    _Float16* rhbuf  = (_Float16*)(ws + offRH);

    prep_pack<<<(3 * 64 * 9 * 64 + 255) / 256, 256, 0, stream>>>(
        Wz, Wr, Wh, Wo, Wpack, Wxpack, Wopack);

    for (int t = 0; t < TT; ++t) {
        gru_gates<<<dim3(HH, BB), 128, 0, stream>>>(
            x, Wpack, Wxpack, bz, br, enc, zbuf, rhbuf, t);
        gru_cand<<<dim3(HH, BB), 128, 0, stream>>>(
            x, Wpack, Wxpack, bh, enc, zbuf, rhbuf, t);
    }

    proj<<<dim3(COUT / 64, HH, BB * TT), 128, 0, stream>>>(enc, Wopack, bo, out);
}
